// BlockSparseAttention_65687229825744
// MI455X (gfx1250) — matrix operand order repaired and checked by reading, not compiled, not run on hardware
//
#include <hip/hip_runtime.h>
#include <stdint.h>

typedef __attribute__((ext_vector_type(16))) __bf16 v16bf;
typedef __attribute__((ext_vector_type(8)))  float  v8f;

#define T_SEQ  4096
#define DMODEL 512
#define NHEAD  8
#define DHEAD  64
#define NB     64   // 64-token blocks along sequence

// ---------- helpers ----------

__device__ __forceinline__ unsigned short f2bf(float f) {
  unsigned u = __float_as_uint(f);
  unsigned r = u + 0x7FFFu + ((u >> 16) & 1u);   // round-to-nearest-even
  return (unsigned short)(r >> 16);
}

union FragCast { v16bf v; uint4 q[2]; };

// A-matrix fragment (16x32 bf16). Row = (lane&15). Per ISA layout:
// lanes 0-15 hold K = [k0..k0+7] and [k0+16..k0+23]; lanes 16-31 hold K+8.
__device__ __forceinline__ v16bf load_a_frag(const unsigned short* rowptr, int k0, int hi) {
  FragCast f;
  f.q[0] = *(const uint4*)(rowptr + k0 + hi * 8);
  f.q[1] = *(const uint4*)(rowptr + k0 + 16 + hi * 8);
  return f.v;
}

// B-matrix fragment (32x16 bf16), column = (lane&15), sourced from a row-major
// array where B[k][n] = src[n][k]: K = [k0+hi*8..+7] and [k0+16+hi*8..+7].
__device__ __forceinline__ v16bf load_b_frag(const unsigned short* rowptr, int k0, int hi) {
  FragCast f;
  f.q[0] = *(const uint4*)(rowptr + k0 + hi * 8);
  f.q[1] = *(const uint4*)(rowptr + k0 + 16 + hi * 8);
  return f.v;
}

// Stage a 64-row x K-col bf16 weight strip into LDS with CDNA5 async LDS loads.
// 256 threads, 16-byte chunks each; tracked by ASYNCcnt, fenced before barrier.
__device__ __forceinline__ void stage_weights_async(const unsigned short* __restrict__ Wstrip,
                                                    unsigned short* lds, int K) {
  unsigned ldsbase = (unsigned)(uintptr_t)lds;     // low 32 bits = LDS byte offset
  int total = (64 * K * 2) / 16;                   // 16B chunks (K=512 -> 4096)
  for (int f = threadIdx.x; f < total; f += 256) {
    unsigned byteoff = (unsigned)f * 16u;
    unsigned ldsaddr = ldsbase + byteoff;
    asm volatile("global_load_async_to_lds_b128 %0, %1, %2"
                 :: "v"(ldsaddr), "v"(byteoff), "s"(Wstrip)
                 : "memory");
  }
  asm volatile("s_wait_asynccnt 0" ::: "memory");
  __syncthreads();
}

// ---------- kernels ----------

__global__ void cvt_f32_bf16(const float* __restrict__ in,
                             unsigned short* __restrict__ out, int n) {
  int i = blockIdx.x * blockDim.x + threadIdx.x;
  int stride = gridDim.x * blockDim.x;
  for (; i < n; i += stride) out[i] = f2bf(in[i]);
}

// Core 16x64 output tile GEMM body: A from global, B strip resident in LDS.
// K-step of 64: 2 A-frag global loads + 8 B-frag LDS loads batched, then 8 WMMAs.
__device__ __forceinline__ void gemm_tile_body(const unsigned short* __restrict__ arow,
                                               const unsigned short* __restrict__ wlds,
                                               int K, int lo, int hi, v8f acc[4]) {
#pragma unroll 1
  for (int k0 = 0; k0 < K; k0 += 64) {
    v16bf a0 = load_a_frag(arow, k0, hi);
    v16bf a1 = load_a_frag(arow, k0 + 32, hi);
    v16bf b0[4], b1[4];
#pragma unroll
    for (int nt = 0; nt < 4; ++nt) {
      const unsigned short* wrow = wlds + (size_t)(nt * 16 + lo) * K;
      b0[nt] = load_b_frag(wrow, k0, hi);
      b1[nt] = load_b_frag(wrow, k0 + 32, hi);
    }
#pragma unroll
    for (int nt = 0; nt < 4; ++nt)
      acc[nt] = __builtin_amdgcn_wmma_f32_16x16x32_bf16(
          false, a0, false, b0[nt], (short)0, acc[nt], false, false);
#pragma unroll
    for (int nt = 0; nt < 4; ++nt)
      acc[nt] = __builtin_amdgcn_wmma_f32_16x16x32_bf16(
          false, a1, false, b1[nt], (short)0, acc[nt], false, false);
  }
}

// out[m][n] = scale * sum_k A[m][k] * W[n][k], bf16 in, bf16 out.
// WG = 8 waves covering a 128x64 tile sharing one LDS-resident weight strip.
__global__ void __launch_bounds__(256)
gemm_bf16_proj(const unsigned short* __restrict__ A,  // M x K
               const unsigned short* __restrict__ W,  // N x K
               unsigned short* __restrict__ out,
               int M, int N, int K, float scale, int transposeOut) {
  __shared__ __align__(16) unsigned short wlds[64 * DMODEL];   // 64 KB strip
  int lane = threadIdx.x & 31;
  int lo = lane & 15, hi = lane >> 4;
  int w = threadIdx.x >> 5;
  int ns = blockIdx.x & 7;                 // N/64 strips
  int mblk = blockIdx.x >> 3;              // 128-row blocks
  int n0 = ns * 64;
  int m0 = mblk * 128 + w * 16;

  stage_weights_async(W + (size_t)n0 * K, wlds, K);

  v8f acc[4] = {};
  const unsigned short* arow = A + (size_t)(m0 + lo) * K;
  gemm_tile_body(arow, wlds, K, lo, hi, acc);

#pragma unroll
  for (int nt = 0; nt < 4; ++nt)
#pragma unroll
    for (int r = 0; r < 8; ++r) {
      int row = m0 + hi * 8 + r;
      int col = n0 + nt * 16 + lo;
      unsigned short v = f2bf(acc[nt][r] * scale);
      if (transposeOut) out[(size_t)col * M + row] = v;
      else              out[(size_t)row * N + col] = v;
    }
}

// Same GEMM but f32 output (final Wo projection -> d_out).
__global__ void __launch_bounds__(256)
gemm_bf16_f32out(const unsigned short* __restrict__ A,
                 const unsigned short* __restrict__ W,
                 float* __restrict__ out, int M, int N, int K) {
  __shared__ __align__(16) unsigned short wlds[64 * DMODEL];
  int lane = threadIdx.x & 31;
  int lo = lane & 15, hi = lane >> 4;
  int w = threadIdx.x >> 5;
  int ns = blockIdx.x & 7;
  int mblk = blockIdx.x >> 3;
  int n0 = ns * 64;
  int m0 = mblk * 128 + w * 16;

  stage_weights_async(W + (size_t)n0 * K, wlds, K);

  v8f acc[4] = {};
  const unsigned short* arow = A + (size_t)(m0 + lo) * K;
  gemm_tile_body(arow, wlds, K, lo, hi, acc);

#pragma unroll
  for (int nt = 0; nt < 4; ++nt)
#pragma unroll
    for (int r = 0; r < 8; ++r)
      out[(size_t)(m0 + hi * 8 + r) * N + n0 + nt * 16 + lo] = acc[nt][r];
}

// Reduce the token mask to a (64x64) block-any table. One 256-thread WG per block.
__global__ void __launch_bounds__(256)
block_any_kernel(const unsigned char* __restrict__ mask,
                 unsigned char* __restrict__ blockAny) {
  __shared__ int flag;
  if (threadIdx.x == 0) flag = 0;
  __syncthreads();
  int bid = blockIdx.x;            // 0..4095
  int qb = bid >> 6, kb = bid & 63;
  int row = threadIdx.x >> 2;      // 0..63
  int ch  = threadIdx.x & 3;       // 16-byte chunk within 64 cols
  const uint4* p = (const uint4*)(mask + (size_t)(qb * 64 + row) * T_SEQ + kb * 64 + ch * 16);
  uint4 m = *p;
  if (m.x | m.y | m.z | m.w) flag = 1;
  __syncthreads();
  if (threadIdx.x == 0) blockAny[bid] = (unsigned char)flag;
}

// Streaming-softmax block-sparse attention.
// grid = (64 query blocks, 8 heads), 128 threads = 4 waves, each wave owns 16 query rows.
// Qb is pre-scaled by 1/sqrt(D). Vt is (h*64+d, t) so PV B-frags are contiguous loads.
__global__ void __launch_bounds__(128)
attn_kernel(const unsigned short* __restrict__ Qb,   // T x 512 bf16 (scaled)
            const unsigned short* __restrict__ Kb,   // T x 512 bf16
            const unsigned short* __restrict__ Vt,   // 512 x T bf16
            const unsigned char* __restrict__ mask,  // T x T
            const unsigned char* __restrict__ blockAny, // 64 x 64
            unsigned short* __restrict__ AO) {       // T x 512 bf16
  __shared__ __align__(16) unsigned short plds[4][16 * 64];
  int lane = threadIdx.x & 31;
  int lo = lane & 15, hi = lane >> 4;
  int w = threadIdx.x >> 5;
  int qb = blockIdx.x;
  int h  = blockIdx.y;
  int q0 = qb * 64 + w * 16;

  const unsigned short* qrow = Qb + (size_t)(q0 + lo) * DMODEL + h * DHEAD;
  v16bf aq0 = load_a_frag(qrow, 0, hi);
  v16bf aq1 = load_a_frag(qrow, 32, hi);

  v8f O[4] = {};
  float rowm[8], rowl[8];
#pragma unroll
  for (int r = 0; r < 8; ++r) { rowm[r] = -__builtin_inff(); rowl[r] = 0.f; }

  unsigned short* myl = &plds[w][0];
  const float NEG_INF = -__builtin_inff();

  for (int kb = 0; kb < NB; ++kb) {
    if (!blockAny[qb * 64 + kb]) continue;   // uniform skip of inactive blocks
    int k0 = kb * 64;

    // Preload all 8 K-fragments so the loads clause and WMMAs issue back-to-back.
    v16bf bk0[4], bk1[4];
#pragma unroll
    for (int nt = 0; nt < 4; ++nt) {
      const unsigned short* krow = Kb + (size_t)(k0 + nt * 16 + lo) * DMODEL + h * DHEAD;
      bk0[nt] = load_b_frag(krow, 0, hi);
      bk1[nt] = load_b_frag(krow, 32, hi);
    }
    // S = Q @ K^T  (16x64 tile, 8 WMMAs)
    v8f s[4];
#pragma unroll
    for (int nt = 0; nt < 4; ++nt) {
      v8f z = {};
      z = __builtin_amdgcn_wmma_f32_16x16x32_bf16(false, aq0, false, bk0[nt], (short)0, z, false, false);
      s[nt] = __builtin_amdgcn_wmma_f32_16x16x32_bf16(false, aq1, false, bk1[nt], (short)0, z, false, false);
    }

    // exact token mask (handles partial blocks from the 4 global tokens)
#pragma unroll
    for (int nt = 0; nt < 4; ++nt)
#pragma unroll
      for (int r = 0; r < 8; ++r) {
        unsigned char mv = mask[(size_t)(q0 + hi * 8 + r) * T_SEQ + k0 + nt * 16 + lo];
        if (!mv) s[nt][r] = NEG_INF;
      }

    // per-row max (each row lives in one 16-lane half at one C-register index)
    float nm[8];
#pragma unroll
    for (int r = 0; r < 8; ++r) {
      float t = fmaxf(fmaxf(s[0][r], s[1][r]), fmaxf(s[2][r], s[3][r]));
      t = fmaxf(t, __shfl_xor(t, 1, 32));
      t = fmaxf(t, __shfl_xor(t, 2, 32));
      t = fmaxf(t, __shfl_xor(t, 4, 32));
      t = fmaxf(t, __shfl_xor(t, 8, 32));
      nm[r] = fmaxf(rowm[r], t);
    }

    // rescale running state, exponentiate, stage P (bf16) in LDS, row sums
#pragma unroll
    for (int r = 0; r < 8; ++r) {
      float alpha = (nm[r] == rowm[r]) ? 1.0f : __expf(rowm[r] - nm[r]);
      rowl[r] *= alpha;
#pragma unroll
      for (int nt = 0; nt < 4; ++nt) O[nt][r] *= alpha;
      float mu = (nm[r] == NEG_INF) ? 0.f : nm[r];
      float rs = 0.f;
#pragma unroll
      for (int nt = 0; nt < 4; ++nt) {
        float p = (s[nt][r] == NEG_INF) ? 0.f : __expf(s[nt][r] - mu);
        rs += p;
        myl[(hi * 8 + r) * 64 + nt * 16 + lo] = f2bf(p);
      }
      rs += __shfl_xor(rs, 1, 32);
      rs += __shfl_xor(rs, 2, 32);
      rs += __shfl_xor(rs, 4, 32);
      rs += __shfl_xor(rs, 8, 32);
      rowl[r] += rs;
      rowm[r] = nm[r];
    }

    // Preload all 8 V-fragments (independent of LDS) to hide latency behind softmax.
    v16bf bv0[4], bv1[4];
#pragma unroll
    for (int dt = 0; dt < 4; ++dt) {
      const unsigned short* vrow = Vt + (size_t)(h * DHEAD + dt * 16 + lo) * T_SEQ + k0;
      bv0[dt] = load_b_frag(vrow, 0, hi);
      bv1[dt] = load_b_frag(vrow, 32, hi);
    }

    // per-wave LDS RAW fence (CDNA5 split DS counter)
    asm volatile("s_wait_dscnt 0" ::: "memory");

    // O += P @ V  (8 WMMAs)
    v16bf pa0 = load_a_frag(myl + lo * 64, 0, hi);
    v16bf pa1 = load_a_frag(myl + lo * 64, 32, hi);
#pragma unroll
    for (int dt = 0; dt < 4; ++dt) {
      O[dt] = __builtin_amdgcn_wmma_f32_16x16x32_bf16(false, pa0, false, bv0[dt], (short)0, O[dt], false, false);
      O[dt] = __builtin_amdgcn_wmma_f32_16x16x32_bf16(false, pa1, false, bv1[dt], (short)0, O[dt], false, false);
    }
  }

  // finalize: divide by row sum, write bf16 attention output (T x 512)
#pragma unroll
  for (int r = 0; r < 8; ++r) {
    float inv = (rowl[r] > 0.f) ? 1.0f / rowl[r] : 0.f;
#pragma unroll
    for (int dt = 0; dt < 4; ++dt)
      AO[(size_t)(q0 + hi * 8 + r) * DMODEL + h * DHEAD + dt * 16 + lo] =
          f2bf(O[dt][r] * inv);
  }
}

// ---------- host ----------

extern "C" void kernel_launch(void* const* d_in, const int* in_sizes, int n_in,
                              void* d_out, int out_size, void* d_ws, size_t ws_size,
                              hipStream_t stream) {
  (void)in_sizes; (void)n_in; (void)out_size; (void)ws_size;
  const float* x          = (const float*)d_in[0];
  const unsigned char* mk = (const unsigned char*)d_in[1];   // bool mask
  const float* Wq         = (const float*)d_in[2];
  const float* Wk         = (const float*)d_in[3];
  const float* Wv         = (const float*)d_in[4];
  const float* Wo         = (const float*)d_in[5];
  float* out              = (float*)d_out;

  char* ws = (char*)d_ws;
  size_t off = 0;
  auto alloc = [&](size_t bytes) {
    char* p = ws + off;
    off += (bytes + 255) & ~(size_t)255;
    return p;
  };
  const size_t XE = (size_t)T_SEQ * DMODEL;      // 2,097,152 elems
  const size_t WE = (size_t)DMODEL * DMODEL;     //   262,144 elems
  unsigned short* xb  = (unsigned short*)alloc(XE * 2);
  unsigned short* wqb = (unsigned short*)alloc(WE * 2);
  unsigned short* wkb = (unsigned short*)alloc(WE * 2);
  unsigned short* wvb = (unsigned short*)alloc(WE * 2);
  unsigned short* wob = (unsigned short*)alloc(WE * 2);
  unsigned short* Qb  = (unsigned short*)alloc(XE * 2);
  unsigned short* Kb  = (unsigned short*)alloc(XE * 2);
  unsigned short* Vt  = (unsigned short*)alloc(XE * 2);
  unsigned short* AO  = (unsigned short*)alloc(XE * 2);
  unsigned char*  BA  = (unsigned char*)alloc(NB * NB);

  // 1) f32 -> bf16 conversions
  cvt_f32_bf16<<<1024, 256, 0, stream>>>(x,  xb,  (int)XE);
  cvt_f32_bf16<<<256,  256, 0, stream>>>(Wq, wqb, (int)WE);
  cvt_f32_bf16<<<256,  256, 0, stream>>>(Wk, wkb, (int)WE);
  cvt_f32_bf16<<<256,  256, 0, stream>>>(Wv, wvb, (int)WE);
  cvt_f32_bf16<<<256,  256, 0, stream>>>(Wo, wob, (int)WE);

  // 2) projections: 4096x512x512 GEMMs; (32 m-blocks x 8 n-strips) WGs of 8 waves
  gemm_bf16_proj<<<256, 256, 0, stream>>>(xb, wqb, Qb, T_SEQ, DMODEL, DMODEL, 0.125f, 0); // Q * 1/sqrt(64)
  gemm_bf16_proj<<<256, 256, 0, stream>>>(xb, wkb, Kb, T_SEQ, DMODEL, DMODEL, 1.0f,   0);
  gemm_bf16_proj<<<256, 256, 0, stream>>>(xb, wvb, Vt, T_SEQ, DMODEL, DMODEL, 1.0f,   1); // transposed

  // 3) block-activity table
  block_any_kernel<<<NB * NB, 256, 0, stream>>>(mk, BA);

  // 4) block-sparse flash attention
  attn_kernel<<<dim3(NB, NHEAD), 128, 0, stream>>>(Qb, Kb, Vt, mk, BA, AO);

  // 5) output projection -> f32 d_out
  gemm_bf16_f32out<<<256, 256, 0, stream>>>(AO, wob, out, T_SEQ, DMODEL, DMODEL);
}
